// InputDrivenCWRNN_2937757630472
// MI455X (gfx1250) — matrix operand order repaired and checked by reading, not compiled, not run on hardware
//
#include <hip/hip_runtime.h>
#include <math.h>

// ---------------------------------------------------------------------------
// InputDrivenCWRNN on MI455X (gfx1250, wave32, WMMA)
//
// Phase 1: U  = x @ W_in^T + b_in   (32768x512 x 512x1024)  -- parallel, WMMA bf16
//          Up = x @ W_ip^T + b_ip
// Phase 2: 512 sequential fused step kernels:
//          h tile staged to LDS via GLOBAL_LOAD_ASYNC_TO_LDS_B128 (ASYNCcnt),
//          acts   = tanh(h @ W_h^T  + U[t])     (WMMA bf16, f32 accum, 2-deep ILP)
//          acts_p = tanh(h @ W_hp^T + Up[t])
//          per-module variance -> gate -> h' = (1-g)*acts + g*h
// ---------------------------------------------------------------------------

typedef __attribute__((ext_vector_type(16))) __bf16 v16bf;
typedef __attribute__((ext_vector_type(8)))  float  v8f;

constexpr int IN_DIM  = 512;
constexpr int OUT_DIM = 1024;
constexpr int NUM_MOD = 16;
constexpr int MOD_SZ  = OUT_DIM / NUM_MOD;   // 64
constexpr int Bb      = 64;
constexpr int Tt      = 512;
constexpr int MROWS   = Bb * Tt;             // 32768

#define DEVINL static __device__ __forceinline__

// Build a v16bf WMMA fragment from two 16-byte chunks (b128 / ds_b128 loads).
DEVINL v16bf pack16(const __bf16* p0, const __bf16* p1) {
  union { uint4 u; __bf16 h[8]; } a, b;
  a.u = *reinterpret_cast<const uint4*>(p0);
  b.u = *reinterpret_cast<const uint4*>(p1);
  v16bf r;
#pragma unroll
  for (int i = 0; i < 8; ++i) { r[i] = a.h[i]; r[i + 8] = b.h[i]; }
  return r;
}

DEVINL v8f wmma_bf16(v16bf a, v16bf b, v8f c) {
  // D = A*B + C, 16x16x32, f32 accum
  return __builtin_amdgcn_wmma_f32_16x16x32_bf16(
      /*neg_a=*/false, a, /*neg_b=*/false, b,
      /*c_mod=*/(short)0, c, /*reuse_a=*/false, /*reuse_b=*/false);
}

// ------------------------------ prep kernels -------------------------------

__global__ void cwrnn_cvt_bf16(const float* __restrict__ src,
                               __bf16* __restrict__ dst, int n) {
  int i = blockIdx.x * 256 + threadIdx.x;
  if (i < n) dst[i] = (__bf16)src[i];
}

__global__ void cwrnn_init_h(float* __restrict__ hf, __bf16* __restrict__ hb, int n) {
  int i = blockIdx.x * 256 + threadIdx.x;
  if (i < n) { hf[i] = 0.0f; hb[i] = (__bf16)0.0f; }
}

// ------------------------- phase 1: input GEMMs ----------------------------
// Grid: (MROWS/128, OUT_DIM/128, 2). Block: 256 threads = 8 waves.
// Each wave computes a 32x64 output patch (2x4 WMMA accumulators).
__global__ void cwrnn_input_gemm(const __bf16* __restrict__ X,
                                 const __bf16* __restrict__ Win,
                                 const __bf16* __restrict__ Wip,
                                 const float*  __restrict__ bin,
                                 const float*  __restrict__ bip,
                                 __bf16* __restrict__ U,
                                 __bf16* __restrict__ Up) {
  const int lane = threadIdx.x & 31;
  const int wave = threadIdx.x >> 5;
  const int wr = wave >> 1;           // 0..3
  const int wc = wave & 1;            // 0..1
  const int rbase = blockIdx.x * 128 + wr * 32;
  const int cbase = blockIdx.y * 128 + wc * 64;

  const __bf16* W   = blockIdx.z ? Wip : Win;
  const float*  bia = blockIdx.z ? bip : bin;
  __bf16*       Uo  = blockIdx.z ? Up  : U;

  const int r = lane & 15;            // row-in-tile (A) / col-in-tile (B,C)
  const int g = lane >> 4;            // lane half -> K sub-group

  v8f acc[2][4];
#pragma unroll
  for (int mt = 0; mt < 2; ++mt)
#pragma unroll
    for (int nt = 0; nt < 4; ++nt) acc[mt][nt] = (v8f){};

  for (int k0 = 0; k0 < IN_DIM; k0 += 32) {
    v16bf afr[2];
#pragma unroll
    for (int mt = 0; mt < 2; ++mt) {
      const __bf16* pa = X + (size_t)(rbase + mt * 16 + r) * IN_DIM + k0 + g * 8;
      afr[mt] = pack16(pa, pa + 16);
    }
    v16bf bfr[4];
#pragma unroll
    for (int nt = 0; nt < 4; ++nt) {
      const __bf16* pb = W + (size_t)(cbase + nt * 16 + r) * IN_DIM + k0 + g * 8;
      bfr[nt] = pack16(pb, pb + 16);
    }
    if (k0 + 32 < IN_DIM) {
      // warm the next K-chunk (emits global_prefetch_b8)
      __builtin_prefetch(X + (size_t)(rbase + r) * IN_DIM + k0 + 32, 0, 3);
      __builtin_prefetch(W + (size_t)(cbase + r) * IN_DIM + k0 + 32, 0, 3);
    }
#pragma unroll
    for (int mt = 0; mt < 2; ++mt)
#pragma unroll
      for (int nt = 0; nt < 4; ++nt)
        acc[mt][nt] = wmma_bf16(afr[mt], bfr[nt], acc[mt][nt]);
  }

  // epilogue: add bias, store bf16
#pragma unroll
  for (int mt = 0; mt < 2; ++mt) {
#pragma unroll
    for (int nt = 0; nt < 4; ++nt) {
      const int col = cbase + nt * 16 + r;
      const float bv = bia[col];
#pragma unroll
      for (int v = 0; v < 8; ++v) {
        const int row = rbase + mt * 16 + v + 8 * g;
        Uo[(size_t)row * OUT_DIM + col] = (__bf16)(acc[mt][nt][v] + bv);
      }
    }
  }
}

// --------------------------- phase 2: scan step ----------------------------
// Grid: (Bb/16, NUM_MOD). Block: 256 threads = 8 waves.
// Waves 0..3: acts tile (16 x 64); waves 4..7: acts_p tile.
// The 16x1024 bf16 h tile (shared A-operand of all 8 waves) is staged into
// LDS with CDNA5 async global->LDS copies, tracked by ASYNCcnt.
__global__ void cwrnn_step(const __bf16* __restrict__ hb_cur,
                           const float*  __restrict__ hf_cur,
                           const __bf16* __restrict__ Wh,
                           const __bf16* __restrict__ Whp,
                           const __bf16* __restrict__ U,
                           const __bf16* __restrict__ Up,
                           const float*  __restrict__ shifts,
                           float*  __restrict__ hf_next,
                           __bf16* __restrict__ hb_next,
                           float*  __restrict__ ys,
                           float*  __restrict__ hout,
                           float*  __restrict__ ps,
                           int t) {
  __shared__ __bf16 s_h[16 * OUT_DIM];        // 32 KB staged h tile
  __shared__ float  s_act[16 * MOD_SZ];
  __shared__ float  s_actp[16 * MOD_SZ];
  __shared__ float  s_gate[16];

  const int tid  = threadIdx.x;
  const int lane = tid & 31;
  const int wave = tid >> 5;
  const int rb   = blockIdx.x;          // row-block: rows 16*rb .. +15 of h
  const int m    = blockIdx.y;          // module index

  // ---- async stage of the h tile: 32 KB = 8 passes x 256 lanes x 16 B ----
  {
    const char* gsrc = (const char*)(hb_cur + (size_t)rb * 16 * OUT_DIM);
    const unsigned lbase = (unsigned)(uintptr_t)(&s_h[0]);   // LDS byte offset
#pragma unroll
    for (int q = 0; q < 8; ++q) {
      const unsigned byteoff = (unsigned)(q * 4096 + tid * 16);
      const unsigned laddr = lbase + byteoff;
      const unsigned long long ga = (unsigned long long)(gsrc + byteoff);
      asm volatile("global_load_async_to_lds_b128 %0, %1, off"
                   :: "v"(laddr), "v"(ga)
                   : "memory");
    }
    asm volatile("s_wait_asynccnt 0x0" ::: "memory");
  }
  __syncthreads();

  const bool isP = wave >= 4;
  const int  nt  = wave & 3;
  const int  col0 = m * MOD_SZ + nt * 16;
  const __bf16* Wmat = isP ? Whp : Wh;
  const __bf16* Umat = isP ? Up  : U;
  float* stile = isP ? s_actp : s_act;

  const int r = lane & 15;
  const int g = lane >> 4;

  const __bf16* arow = &s_h[(size_t)r * OUT_DIM];             // LDS (ds_load)
  const __bf16* brow = Wmat + (size_t)(col0 + r) * OUT_DIM;   // global (L2-hot)

  // two independent accumulator chains -> 2x WMMA ILP on the critical path
  v8f acc0 = (v8f){};
  v8f acc1 = (v8f){};
  for (int k0 = 0; k0 < OUT_DIM; k0 += 64) {
    v16bf a0 = pack16(arow + k0 + g * 8,       arow + k0 + 16 + g * 8);
    v16bf b0 = pack16(brow + k0 + g * 8,       brow + k0 + 16 + g * 8);
    v16bf a1 = pack16(arow + k0 + 32 + g * 8,  arow + k0 + 48 + g * 8);
    v16bf b1 = pack16(brow + k0 + 32 + g * 8,  brow + k0 + 48 + g * 8);
    if (k0 + 64 < OUT_DIM)
      __builtin_prefetch(brow + k0 + 64, 0, 3);
    acc0 = wmma_bf16(a0, b0, acc0);
    acc1 = wmma_bf16(a1, b1, acc1);
  }
  v8f acc = acc0 + acc1;

  // add precomputed input projection, tanh, stash in LDS
#pragma unroll
  for (int v = 0; v < 8; ++v) {
    const int row   = v + 8 * g;
    const int bglob = rb * 16 + row;
    const int o     = col0 + r;
    const float pre = acc[v] + (float)Umat[((size_t)bglob * Tt + t) * OUT_DIM + o];
    stile[row * MOD_SZ + nt * 16 + r] = tanhf(pre);
  }
  __syncthreads();

  // per-row variance over the 64-wide module (ddof=1) -> gate
  {
    const int row = tid >> 4;           // 0..15
    const int j   = tid & 15;
    float s = 0.0f, s2 = 0.0f;
#pragma unroll
    for (int q = 0; q < 4; ++q) {
      const float v = s_actp[row * MOD_SZ + j + q * 16];
      s += v; s2 += v * v;
    }
#pragma unroll
    for (int off = 8; off > 0; off >>= 1) {
      s  += __shfl_xor(s,  off, 16);
      s2 += __shfl_xor(s2, off, 16);
    }
    if (j == 0) {
      const float mean = s * (1.0f / MOD_SZ);
      const float var  = (s2 - (float)MOD_SZ * mean * mean) * (1.0f / (MOD_SZ - 1));
      const float gate = 0.5f * (sinf((float)t * var + shifts[m]) + 1.0f);
      s_gate[row] = gate;
      const int bglob = rb * 16 + row;
      ps[((size_t)bglob * Tt + t) * NUM_MOD + m] = var;
    }
  }
  __syncthreads();

  // blend and emit h', ys[t]
#pragma unroll
  for (int q = 0; q < 4; ++q) {
    const int idx   = tid + q * 256;
    const int row   = idx >> 6;
    const int c     = idx & 63;
    const int bglob = rb * 16 + row;
    const int o     = m * MOD_SZ + c;
    const float a   = s_act[row * MOD_SZ + c];
    const float hv  = hf_cur[(size_t)bglob * OUT_DIM + o];
    const float gte = s_gate[row];
    const float y   = (1.0f - gte) * a + gte * hv;
    hf_next[(size_t)bglob * OUT_DIM + o] = y;
    hb_next[(size_t)bglob * OUT_DIM + o] = (__bf16)y;
    ys[((size_t)bglob * Tt + t) * OUT_DIM + o] = y;
    hout[(size_t)bglob * OUT_DIM + o] = y;   // last step's write = final h
  }
}

// ------------------------------- launcher ----------------------------------

extern "C" void kernel_launch(void* const* d_in, const int* in_sizes, int n_in,
                              void* d_out, int out_size, void* d_ws, size_t ws_size,
                              hipStream_t stream) {
  const float* x      = (const float*)d_in[0];   // [B, T, IN]
  const float* W_in   = (const float*)d_in[1];   // [OUT, IN]
  const float* b_in   = (const float*)d_in[2];   // [OUT]
  const float* W_h    = (const float*)d_in[3];   // [OUT, OUT]
  const float* W_ip   = (const float*)d_in[4];   // [OUT, IN]
  const float* b_ip   = (const float*)d_in[5];   // [OUT]
  const float* W_hp   = (const float*)d_in[6];   // [OUT, OUT]
  const float* shifts = (const float*)d_in[7];   // [M]

  float* out  = (float*)d_out;
  float* ys   = out;                                       // [B, T, OUT]
  float* hout = ys + (size_t)Bb * Tt * OUT_DIM;            // [B, OUT]
  float* ps   = hout + (size_t)Bb * OUT_DIM;               // [B, T, M]

  // workspace carve (256B aligned slabs)
  char* w = (char*)d_ws;
  size_t off = 0;
  auto alloc = [&](size_t bytes) -> void* {
    void* p = w + off;
    off += (bytes + 255) & ~(size_t)255;
    return p;
  };
  __bf16* Xb   = (__bf16*)alloc((size_t)MROWS * IN_DIM * 2);
  __bf16* Winb = (__bf16*)alloc((size_t)OUT_DIM * IN_DIM * 2);
  __bf16* Wipb = (__bf16*)alloc((size_t)OUT_DIM * IN_DIM * 2);
  __bf16* Whb  = (__bf16*)alloc((size_t)OUT_DIM * OUT_DIM * 2);
  __bf16* Whpb = (__bf16*)alloc((size_t)OUT_DIM * OUT_DIM * 2);
  __bf16* Ub   = (__bf16*)alloc((size_t)MROWS * OUT_DIM * 2);
  __bf16* Upb  = (__bf16*)alloc((size_t)MROWS * OUT_DIM * 2);
  float*  hf0  = (float*) alloc((size_t)Bb * OUT_DIM * 4);
  float*  hf1  = (float*) alloc((size_t)Bb * OUT_DIM * 4);
  __bf16* hb0  = (__bf16*)alloc((size_t)Bb * OUT_DIM * 2);
  __bf16* hb1  = (__bf16*)alloc((size_t)Bb * OUT_DIM * 2);

  // phase 0: bf16 conversions + h0 = 0
  {
    int n = MROWS * IN_DIM;
    cwrnn_cvt_bf16<<<(n + 255) / 256, 256, 0, stream>>>(x, Xb, n);
    n = OUT_DIM * IN_DIM;
    cwrnn_cvt_bf16<<<(n + 255) / 256, 256, 0, stream>>>(W_in, Winb, n);
    cwrnn_cvt_bf16<<<(n + 255) / 256, 256, 0, stream>>>(W_ip, Wipb, n);
    n = OUT_DIM * OUT_DIM;
    cwrnn_cvt_bf16<<<(n + 255) / 256, 256, 0, stream>>>(W_h, Whb, n);
    cwrnn_cvt_bf16<<<(n + 255) / 256, 256, 0, stream>>>(W_hp, Whpb, n);
    n = Bb * OUT_DIM;
    cwrnn_init_h<<<(n + 255) / 256, 256, 0, stream>>>(hf0, hb0, n);
  }

  // phase 1: hoisted input projections (both weight sets, via blockIdx.z)
  cwrnn_input_gemm<<<dim3(MROWS / 128, OUT_DIM / 128, 2), 256, 0, stream>>>(
      Xb, Winb, Wipb, b_in, b_ip, Ub, Upb);

  // phase 2: sequential scan, ping-pong h
  for (int t = 0; t < Tt; ++t) {
    const bool even = (t & 1) == 0;
    cwrnn_step<<<dim3(Bb / 16, NUM_MOD), 256, 0, stream>>>(
        even ? hb0 : hb1, even ? hf0 : hf1,
        Whb, Whpb, Ub, Upb, shifts,
        even ? hf1 : hf0, even ? hb1 : hb0,
        ys, hout, ps, t);
  }
}
